// Attention_34119220199768
// MI455X (gfx1250) — matrix operand order repaired and checked by reading, not compiled, not run on hardware
//
#include <hip/hip_runtime.h>
#include <hip/hip_bf16.h>
#include <cstdint>
#include <cstddef>

#define DEV_INLINE __device__ __forceinline__

typedef __attribute__((ext_vector_type(16))) __bf16 v16bf;
typedef __attribute__((ext_vector_type(4)))  __bf16 v4bf;
typedef __attribute__((ext_vector_type(8)))  float  v8f;
typedef __attribute__((ext_vector_type(4)))  float  v4f;
typedef __attribute__((ext_vector_type(4)))  unsigned int v4u;

constexpr int B_    = 16;
constexpr int S_    = 512;
constexpr int W_    = 8;
constexpr int DQ_   = 256;
constexpr int DIN_  = 512;
constexpr int DOUT_ = 512;

constexpr int KC   = 256;   // K chunk staged in LDS
constexpr int KPAD = 264;   // +8 bf16 pad -> 132-dword column stride (bank rotate 4)

// ---------------------------------------------------------------- helpers

DEV_INLINE __bf16 f32_to_bf16(float f) {
  unsigned int u = __builtin_bit_cast(unsigned int, f);
  u += 0x7FFFu + ((u >> 16) & 1u);           // round-to-nearest-even
  unsigned short h = (unsigned short)(u >> 16);
  return __builtin_bit_cast(__bf16, h);
}

union Frag { v16bf bf; v4u q[2]; };

// A fragment: 16x32 bf16, lane = M (mod 16), half = lane>>4.
// Lane covers K in [8h, 8h+8) (V0..3) and [16+8h, 16+8h+8) (V4..7):
// two contiguous 16B chunks -> 2x global_load_b128.
DEV_INLINE Frag load_frag_a(const __bf16* base, int ld, int row, int k0, int half) {
  const v4u* p = (const v4u*)(base + (size_t)row * ld + k0 + half * 8);
  Frag f; f.q[0] = p[0]; f.q[1] = p[2];
  return f;
}

// Stage one 64(N) x 256(K) bf16 B tile into LDS. 256 threads, each copies
// 64 contiguous elements (8x b128 load + ds_store_b128).
DEV_INLINE void stage_b(const __bf16* __restrict__ Bt, int ldb, int n0, int k0,
                        __bf16* smemB) {
  const int row = threadIdx.x >> 2;      // 0..63 -> N within tile
  const int seg = threadIdx.x & 3;       // 0..3  -> 64-element K segment
  const __bf16* src = Bt + (size_t)(n0 + row) * ldb + k0 + seg * 64;
  __bf16* dst = smemB + row * KPAD + seg * 64;
#pragma unroll
  for (int i = 0; i < 64; i += 8)
    *(v4u*)(dst + i) = *(const v4u*)(src + i);
}

// Wave computes a 32(M) x 64(N) strip: 2 A-frags (global) x 4 B-frags (LDS)
// -> 8 WMMAs per 32-wide K step. Block = 8 waves -> 256x64 tile.
DEV_INLINE void chunk_mma(const __bf16* __restrict__ A, int lda,
                          const __bf16* smemB, int k0, int m0, v8f acc[2][4]) {
  const int lane = threadIdx.x & 31;
  const int r16  = lane & 15;
  const int half = lane >> 4;
  for (int k = 0; k < KC; k += 32) {
    Frag fa0 = load_frag_a(A, lda, m0 +      r16, k0 + k, half);
    Frag fa1 = load_frag_a(A, lda, m0 + 16 + r16, k0 + k, half);
#pragma unroll
    for (int t = 0; t < 4; ++t) {
      const v4u* p = (const v4u*)(smemB + (t * 16 + r16) * KPAD + k + half * 8);
      Frag fb; fb.q[0] = p[0]; fb.q[1] = p[2];
      acc[0][t] = __builtin_amdgcn_wmma_f32_16x16x32_bf16(false, fa0.bf, false, fb.bf, (short)0, acc[0][t], false, false);
      acc[1][t] = __builtin_amdgcn_wmma_f32_16x16x32_bf16(false, fa1.bf, false, fb.bf, (short)0, acc[1][t], false, false);
    }
  }
}

DEV_INLINE void run_gemm_core(const __bf16* A, int lda, const __bf16* Bt, int ldb,
                              int K, int m0, int n0, __bf16* smemB, v8f acc[2][4]) {
  for (int kk = 0; kk < K; kk += KC) {
    stage_b(Bt, ldb, n0, kk, smemB);
    __syncthreads();
    chunk_mma(A, lda, smemB, kk, m0, acc);
    __syncthreads();
  }
}

// ---------------------------------------------------------------- kernels

// f32 -> bf16 elementwise (n multiple of 4)
__global__ void cvt_f32_bf16(const float* __restrict__ in, __bf16* __restrict__ out, int n) {
  int i = (blockIdx.x * blockDim.x + threadIdx.x) * 4;
  if (i < n) {
    v4f v = *(const v4f*)(in + i);
    v4bf o;
    o[0] = f32_to_bf16(v[0]); o[1] = f32_to_bf16(v[1]);
    o[2] = f32_to_bf16(v[2]); o[3] = f32_to_bf16(v[3]);
    *(v4bf*)(out + i) = o;
  }
}

// W[K,N] f32 -> WT[N,K] bf16
__global__ void cvt_transpose(const float* __restrict__ in, __bf16* __restrict__ out,
                              int K, int N) {
  int i = blockIdx.x * blockDim.x + threadIdx.x;
  if (i < K * N) {
    int k = i / N, n = i % N;
    out[(size_t)n * K + k] = f32_to_bf16(in[i]);
  }
}

// words_mask[b,s] = any_w content_mask[b,s,w]   (W_ == 8 bytes per row)
__global__ void make_wmask(const unsigned char* __restrict__ cm,
                           unsigned char* __restrict__ wm, int n) {
  int i = blockIdx.x * blockDim.x + threadIdx.x;
  if (i < n) {
    const unsigned long long* p = (const unsigned long long*)cm;
    wm[i] = p[i] ? 1 : 0;
  }
}

// C[M,N] = A[M,K] * Bt[N,K]^T + bias;  out bf16 or f32, row stride ldc.
template <bool F32OUT>
__global__ void gemm_nt_bias(const __bf16* __restrict__ A, int lda,
                             const __bf16* __restrict__ Bt, int ldb,
                             const float* __restrict__ bias,
                             void* __restrict__ out, int ldc, int K) {
  __shared__ __bf16 smemB[64 * KPAD];
  const int wave = threadIdx.x >> 5, lane = threadIdx.x & 31;
  const int m0 = blockIdx.y * 256 + wave * 32;
  const int n0 = blockIdx.x * 64;
  v8f acc[2][4] = {};
  run_gemm_core(A, lda, Bt, ldb, K, m0, n0, smemB, acc);
  const int half = lane >> 4, nloc = lane & 15;
#pragma unroll
  for (int mi = 0; mi < 2; ++mi) {
    const int mbase = m0 + mi * 16 + 8 * half;
#pragma unroll
    for (int t = 0; t < 4; ++t) {
      int n = n0 + t * 16 + nloc;
      float bv = bias ? bias[n] : 0.f;
      if constexpr (F32OUT) {
        float* o = (float*)out;
#pragma unroll
        for (int r = 0; r < 8; ++r) o[(size_t)(mbase + r) * ldc + n] = acc[mi][t][r] + bv;
      } else {
        __bf16* o = (__bf16*)out;
#pragma unroll
        for (int r = 0; r < 8; ++r) o[(size_t)(mbase + r) * ldc + n] = f32_to_bf16(acc[mi][t][r] + bv);
      }
    }
  }
}

// v projection with transposed store: vT[b][z][k] = (data[b] @ Wv + bv)[k][z]
__global__ void gemm_v_trans(const __bf16* __restrict__ db, const __bf16* __restrict__ WvT,
                             const float* __restrict__ bias, __bf16* __restrict__ vT) {
  __shared__ __bf16 smemB[64 * KPAD];
  const int b = blockIdx.z;
  const __bf16* A = db + (size_t)b * S_ * DIN_;
  const int wave = threadIdx.x >> 5, lane = threadIdx.x & 31;
  const int m0 = blockIdx.y * 256 + wave * 32;
  const int n0 = blockIdx.x * 64;
  v8f acc[2][4] = {};
  run_gemm_core(A, DIN_, WvT, DIN_, DIN_, m0, n0, smemB, acc);
  const int half = lane >> 4, nloc = lane & 15;
  __bf16* ob = vT + (size_t)b * DOUT_ * S_;
#pragma unroll
  for (int mi = 0; mi < 2; ++mi) {
    const int mbase = m0 + mi * 16 + 8 * half;
#pragma unroll
    for (int t = 0; t < 4; ++t) {
      int n = n0 + t * 16 + nloc;
      float bv = bias[n];
      __bf16* dst = ob + (size_t)n * S_ + mbase;   // contiguous in m
#pragma unroll
      for (int r = 0; r < 8; ++r) dst[r] = f32_to_bf16(acc[mi][t][r] + bv);
    }
  }
}

// scores[b,q,k,w] = (qs[b,q,w,:] . k[b,k,:]) * scale, masked to -1e9
__global__ void gemm_scores(const __bf16* __restrict__ qsb, const __bf16* __restrict__ kb,
                            const unsigned char* __restrict__ cm,
                            const unsigned char* __restrict__ wm,
                            float* __restrict__ attf, float scale) {
  __shared__ __bf16 smemB[64 * KPAD];
  const int b = blockIdx.z;
  const __bf16* A  = qsb + (size_t)b * S_ * W_ * DIN_;  // [S*W, DIN]
  const __bf16* Bt = kb  + (size_t)b * S_ * DIN_;       // [S, DIN] (NT)
  const int wave = threadIdx.x >> 5, lane = threadIdx.x & 31;
  const int m0 = blockIdx.y * 256 + wave * 32;
  const int n0 = blockIdx.x * 64;
  v8f acc[2][4] = {};
  run_gemm_core(A, DIN_, Bt, DIN_, DIN_, m0, n0, smemB, acc);
  const int half = lane >> 4, nloc = lane & 15;
#pragma unroll
  for (int mi = 0; mi < 2; ++mi) {
    const int mbase = m0 + mi * 16 + 8 * half;
#pragma unroll
    for (int t = 0; t < 4; ++t) {
      int n = n0 + t * 16 + nloc;                       // key index
      bool kmask = wm[b * S_ + n] != 0;
#pragma unroll
      for (int r = 0; r < 8; ++r) {
        int m = mbase + r;
        int q = m >> 3, w = m & 7;
        bool ok = kmask && (cm[((size_t)b * S_ + q) * W_ + w] != 0);
        float v = ok ? acc[mi][t][r] * scale : -1e9f;
        attf[(((size_t)b * S_ + q) * S_ + n) * W_ + w] = v;
      }
    }
  }
}

// softmax over k; one wave per (b,q,w) row; also writes bf16 copy [b,w,q,k]
__global__ void softmax_rows(float* __restrict__ attf, __bf16* __restrict__ attb) {
  const int lane = threadIdx.x & 31;
  const size_t row = (size_t)blockIdx.x * 8 + (threadIdx.x >> 5); // < B*S*W
  const int b = (int)(row >> 12);
  const int q = (int)((row >> 3) & (S_ - 1));
  const int w = (int)(row & 7);
  float* p = attf + (((size_t)b * S_ + q) * S_) * W_ + w;
  float vals[16];
  float mx = -3.0e38f;
#pragma unroll
  for (int i = 0; i < 16; ++i) {
    vals[i] = p[(size_t)(lane + 32 * i) * W_];
    mx = fmaxf(mx, vals[i]);
  }
#pragma unroll
  for (int off = 16; off >= 1; off >>= 1) mx = fmaxf(mx, __shfl_xor(mx, off, 32));
  float sum = 0.f;
#pragma unroll
  for (int i = 0; i < 16; ++i) { vals[i] = __expf(vals[i] - mx); sum += vals[i]; }
#pragma unroll
  for (int off = 16; off >= 1; off >>= 1) sum += __shfl_xor(sum, off, 32);
  float inv = 1.f / sum;
  __bf16* ob = attb + (((size_t)b * W_ + w) * S_ + q) * S_;
#pragma unroll
  for (int i = 0; i < 16; ++i) {
    float a = vals[i] * inv;
    p[(size_t)(lane + 32 * i) * W_] = a;
    ob[lane + 32 * i] = f32_to_bf16(a);
  }
}

// z[b,q,w,:] = att_bf16[b,w,q,:] @ v[b]   (NT with vT), 128 batched GEMMs
__global__ void gemm_z(const __bf16* __restrict__ attb, const __bf16* __restrict__ vT,
                       __bf16* __restrict__ zb) {
  __shared__ __bf16 smemB[64 * KPAD];
  const int zi = blockIdx.z;           // b*8 + w
  const int b = zi >> 3, w = zi & 7;
  const __bf16* A  = attb + (size_t)zi * S_ * S_;
  const __bf16* Bt = vT   + (size_t)b * DOUT_ * S_;
  const int wave = threadIdx.x >> 5, lane = threadIdx.x & 31;
  const int m0 = blockIdx.y * 256 + wave * 32;
  const int n0 = blockIdx.x * 64;
  v8f acc[2][4] = {};
  run_gemm_core(A, S_, Bt, S_, S_, m0, n0, smemB, acc);
  const int half = lane >> 4, nloc = lane & 15;
  __bf16* out = zb + ((size_t)b * S_ * W_ + w) * DOUT_;  // row q stride = W_*DOUT_
#pragma unroll
  for (int mi = 0; mi < 2; ++mi) {
    const int mbase = m0 + mi * 16 + 8 * half;
#pragma unroll
    for (int t = 0; t < 4; ++t) {
      int n = n0 + t * 16 + nloc;
#pragma unroll
      for (int r = 0; r < 8; ++r)
        out[(size_t)(mbase + r) * (W_ * DOUT_) + n] = f32_to_bf16(acc[mi][t][r]);
    }
  }
}

// ---------------------------------------------------------------- launch

extern "C" void kernel_launch(void* const* d_in, const int* in_sizes, int n_in,
                              void* d_out, int out_size, void* d_ws, size_t ws_size,
                              hipStream_t stream) {
  const float* query = (const float*)d_in[0];
  const float* data  = (const float*)d_in[1];
  const unsigned char* cmask = (const unsigned char*)d_in[2];
  const float* Wq = (const float*)d_in[3];  const float* bq = (const float*)d_in[4];
  const float* Wk = (const float*)d_in[5];  const float* bk = (const float*)d_in[6];
  const float* Wv = (const float*)d_in[7];  const float* bv = (const float*)d_in[8];
  const float* Wz = (const float*)d_in[9];  const float* bz = (const float*)d_in[10];

  const size_t n_q   = (size_t)B_ * S_ * W_ * DQ_;   // 16,777,216
  const size_t n_d   = (size_t)B_ * S_ * DIN_;       //  4,194,304
  const size_t n_qs  = (size_t)B_ * S_ * W_ * DIN_;  // 33,554,432
  const size_t n_att = (size_t)B_ * S_ * S_ * W_;    // 33,554,432
  const size_t n_zs  = (size_t)B_ * S_ * W_ * DOUT_; // 33,554,432

  float* zs_out  = (float*)d_out;
  float* att_out = zs_out + n_zs;

  char* ws = (char*)d_ws;
  size_t off = 0;
  auto take = [&](size_t bytes) -> char* {
    char* p = ws + off;
    off = (off + bytes + 255) & ~(size_t)255;
    return p;
  };
  __bf16* qb   = (__bf16*)take(n_q * 2);
  __bf16* db   = (__bf16*)take(n_d * 2);
  __bf16* WqT  = (__bf16*)take((size_t)DQ_ * DIN_ * 2);
  __bf16* WkT  = (__bf16*)take((size_t)DIN_ * DIN_ * 2);
  __bf16* WvT  = (__bf16*)take((size_t)DIN_ * DOUT_ * 2);
  __bf16* WzT  = (__bf16*)take((size_t)DIN_ * DOUT_ * 2);
  __bf16* qsb  = (__bf16*)take(n_qs * 2);
  __bf16* kb   = (__bf16*)take(n_d * 2);
  __bf16* vT   = (__bf16*)take(n_d * 2);
  __bf16* attb = (__bf16*)take(n_att * 2);
  __bf16* zb   = (__bf16*)take(n_zs * 2);
  unsigned char* wmask = (unsigned char*)take((size_t)B_ * S_);

  const dim3 blk(256, 1, 1);

  // 1) conversions
  cvt_f32_bf16<<<dim3((unsigned)(n_q / 1024)), blk, 0, stream>>>(query, qb, (int)n_q);
  cvt_f32_bf16<<<dim3((unsigned)(n_d / 1024)), blk, 0, stream>>>(data, db, (int)n_d);
  cvt_transpose<<<dim3((DQ_ * DIN_ + 255) / 256), blk, 0, stream>>>(Wq, WqT, DQ_, DIN_);
  cvt_transpose<<<dim3((DIN_ * DIN_ + 255) / 256), blk, 0, stream>>>(Wk, WkT, DIN_, DIN_);
  cvt_transpose<<<dim3((DIN_ * DOUT_ + 255) / 256), blk, 0, stream>>>(Wv, WvT, DIN_, DOUT_);
  cvt_transpose<<<dim3((DIN_ * DOUT_ + 255) / 256), blk, 0, stream>>>(Wz, WzT, DIN_, DOUT_);
  make_wmask<<<dim3((B_ * S_ + 255) / 256), blk, 0, stream>>>(cmask, wmask, B_ * S_);

  // 2) qs = query @ Wq + bq  (M=65536, N=512, K=256) -> bf16
  gemm_nt_bias<false><<<dim3(DIN_ / 64, (B_ * S_ * W_) / 256, 1), blk, 0, stream>>>(
      qb, DQ_, WqT, DQ_, bq, qsb, DIN_, DQ_);
  // 3) k = data @ Wk + bk  (M=8192, N=512, K=512) -> bf16
  gemm_nt_bias<false><<<dim3(DIN_ / 64, (B_ * S_) / 256, 1), blk, 0, stream>>>(
      db, DIN_, WkT, DIN_, bk, kb, DIN_, DIN_);
  // 4) vT[b][z][k] = (data[b] @ Wv + bv)^T   (16 batched, transposed store)
  gemm_v_trans<<<dim3(DOUT_ / 64, S_ / 256, B_), blk, 0, stream>>>(db, WvT, bv, vT);

  // 5) masked scores -> att f32 output region
  const float scale = 0.044194173824159216f;  // 1/sqrt(512)
  gemm_scores<<<dim3(S_ / 64, (S_ * W_) / 256, B_), blk, 0, stream>>>(
      qsb, kb, cmask, wmask, att_out, scale);

  // 6) softmax over k (wave per row), writes f32 att + bf16 copy [b,w,q,k]
  softmax_rows<<<dim3((B_ * S_ * W_) / 8), blk, 0, stream>>>(att_out, attb);

  // 7) z = att @ v   (128 batched GEMMs, M=N=K=512) -> bf16 [b,q,w,z]
  gemm_z<<<dim3(DOUT_ / 64, S_ / 256, B_ * W_), blk, 0, stream>>>(attb, vT, zb);

  // 8) zs = z @ Wz + bz  (M=65536, N=512, K=512) -> f32 output
  gemm_nt_bias<true><<<dim3(DOUT_ / 64, (B_ * S_ * W_) / 256, 1), blk, 0, stream>>>(
      zb, DIN_, WzT, DIN_, bz, zs_out, DOUT_, DIN_);
}
